// CBCForecasterV2_38062000177699
// MI455X (gfx1250) — compile-verified, matrix operand order checked by reading, not run on hardware
//
#include <hip/hip_runtime.h>
#include <hip/hip_bf16.h>

// ---------------- model dimensions ----------------
#define BTOT  32768
#define T_IN  3
#define F_IN  6
#define NBASE 10
#define H     256
#define BD    128
#define T_OUT 5
#define DIN   390
#define DINP  416   // DIN padded to multiple of 32
#define G4H   1024  // 4*H gate width

typedef __attribute__((ext_vector_type(16))) _Float16 v16h;
typedef __attribute__((ext_vector_type(8)))  float    v8f;

union HFrag { v16h v; _Float16 h[16]; };

// Compiler-only barrier: forces weight fragments to be re-loaded (from L2)
// each sequential iteration instead of being LICM-hoisted and spilled to scratch.
__device__ __forceinline__ void opaque_iter() { asm volatile("" ::: "memory"); }

// f16 weight workspace offsets (in halves).
// Fragment-major layout: for logical [Kpad][N], fragment f = kc*(N/16)+nt holds
// 512 halves as [lane][e] so each lane reads 32B contiguous (2x global_load_b128).
static constexpr size_t OFF_IH0 = 0;                       // 32   x 1024
static constexpr size_t OFF_HH0 = OFF_IH0 + 32u   * 1024;  // 256  x 1024
static constexpr size_t OFF_IH1 = OFF_HH0 + 256u  * 1024;  // 256  x 1024
static constexpr size_t OFF_HH1 = OFF_IH1 + 256u  * 1024;  // 256  x 1024
static constexpr size_t OFF_2A  = OFF_HH1 + 256u  * 1024;  // 256  x 256
static constexpr size_t OFF_1A  = OFF_2A  + 256u  * 256;   // 256  x 256
static constexpr size_t OFF_DIH = OFF_1A  + 256u  * 256;   // 416  x 1024
static constexpr size_t OFF_DHH = OFF_DIH + 416u  * 1024;  // 256  x 1024
static constexpr size_t OFF_O1  = OFF_DHH + 256u  * 1024;  // 256  x 128
static constexpr size_t WS_HALVES = OFF_O1 + 256u * 128;

// K-index of element e within a 16-half A/B fragment for lane L
__device__ __forceinline__ int kidx(int e, int lane) {
  return ((e >> 3) << 4) + ((lane >> 4) << 3) + (e & 7);
}

// A fragment (16 rows x 32 k) from a row-major [16][K] f16 LDS buffer
// (per lane: two contiguous 8-half runs -> 2x ds_load_b128)
__device__ __forceinline__ v16h load_a_lds(const _Float16* buf, int K, int kc, int lane) {
  HFrag a;
  const int m = lane & 15, kb = kc * 32;
#pragma unroll
  for (int e = 0; e < 16; ++e) a.h[e] = buf[m * K + kb + kidx(e, lane)];
  return a.v;
}

// B fragment from fragment-major swizzled global weights: 32B contiguous per lane
__device__ __forceinline__ v16h load_b_glb(const _Float16* wt, int N, int kc, int n0, int lane) {
  const size_t frag = (size_t)kc * (N >> 4) + (n0 >> 4);
  return *(const v16h*)(wt + frag * 512 + lane * 16);
}

__device__ __forceinline__ float sigf(float x) { return 1.f / (1.f + __expf(-x)); }

// accumulate the 4-gate GEMM z += A @ Wt for this wave's 32-column slice
__device__ __forceinline__ void gemm_gates(v8f acc[2][4], const _Float16* abuf, int K,
                                           const _Float16* wt, int nb, int lane) {
  const int KC = K / 32;
  for (int kc = 0; kc < KC; ++kc) {
    v16h a = load_a_lds(abuf, K, kc, lane);
#pragma unroll
    for (int cc = 0; cc < 2; ++cc) {
#pragma unroll
      for (int g = 0; g < 4; ++g) {
        v16h b = load_b_glb(wt, G4H, kc, g * H + nb + cc * 16, lane);
        acc[cc][g] = __builtin_amdgcn_wmma_f32_16x16x32_f16(
            false, a, false, b, (short)0, acc[cc][g], false, false);
      }
    }
  }
}

// LSTM elementwise: consume gate accumulators, update register c, write h (f16)
__device__ __forceinline__ void lstm_elem(v8f acc[2][4], float c_reg[2][8], int nb, int lane,
                                          const float* b_ih, const float* b_hh,
                                          _Float16* s_h, _Float16* ybuf) {
  const int hi = (lane >> 4) << 3;
#pragma unroll
  for (int cc = 0; cc < 2; ++cc) {
    const int col = nb + cc * 16 + (lane & 15);
    const float bi = b_ih[        col] + b_hh[        col];
    const float bf = b_ih[H     + col] + b_hh[H     + col];
    const float bg = b_ih[2 * H + col] + b_hh[2 * H + col];
    const float bo = b_ih[3 * H + col] + b_hh[3 * H + col];
#pragma unroll
    for (int v = 0; v < 8; ++v) {
      const int row = hi + v;
      const float iv = sigf(acc[cc][0][v] + bi);
      const float fv = sigf(acc[cc][1][v] + bf);
      const float gv = tanhf(acc[cc][2][v] + bg);
      const float ov = sigf(acc[cc][3][v] + bo);
      const float c = fv * c_reg[cc][v] + iv * gv;
      c_reg[cc][v] = c;
      const float h = ov * tanhf(c);
      s_h[row * H + col] = (_Float16)h;
      if (ybuf) ybuf[row * H + col] = (_Float16)h;
    }
  }
}

// ---------------- weight prep ----------------
// fp32 [N][K] (model layout W[n][k]) -> f16 fragment-major swizzle, zero-padded in K.
__global__ void prep_wt_kernel(const float* __restrict__ W, _Float16* __restrict__ Wt,
                               int N, int K, int Kpad) {
  int idx = blockIdx.x * 256 + threadIdx.x;
  if (idx >= Kpad * N) return;
  const int frag   = idx >> 9;        // 512 halves per fragment
  const int within = idx & 511;
  const int lane   = within >> 4;
  const int e      = within & 15;
  const int ntiles = N >> 4;
  const int kc     = frag / ntiles;
  const int nt     = frag - kc * ntiles;
  const int k      = kc * 32 + kidx(e, lane);
  const int n      = nt * 16 + (lane & 15);
  Wt[idx] = (k < K) ? (_Float16)W[(size_t)n * K + k] : (_Float16)0.f;
}

// ---------------- fused persistent forecaster: one 16-row batch tile / block ----------------
__global__ __launch_bounds__(256)
void cbc_forecast_kernel(const float* __restrict__ x, const float* __restrict__ baseline,
                         const float* __restrict__ b_ih0, const float* __restrict__ b_hh0,
                         const float* __restrict__ b_ih1, const float* __restrict__ b_hh1,
                         const float* __restrict__ Wb,    const float* __restrict__ bb,
                         const float* __restrict__ ln_g,  const float* __restrict__ ln_b,
                         const float* __restrict__ b1a,   const float* __restrict__ b2a,
                         const float* __restrict__ va,
                         const float* __restrict__ bd_ih, const float* __restrict__ bd_hh,
                         const float* __restrict__ bo1,   const float* __restrict__ Wo2,
                         const float* __restrict__ bo2,
                         const _Float16* __restrict__ ws, float* __restrict__ out) {
  // ---- LDS (static, ~61 KB total) ----
  __shared__ __align__(16) char s_union[24576];       // phase A: y0[3][16][256] f16
                                                      // phase B/C: din[16][416] f16 + scratch
  __shared__ _Float16 s_y1[3 * 16 * H];               // 24 KB encoder outputs
  __shared__ _Float16 s_h[16 * H];                    // 8 KB current hidden (WMMA A source)
  __shared__ _Float16 s_x[3 * 16 * 32];               // 3 KB zero-padded input tiles
  __shared__ float s_sc[16 * 3];                      // attention scores
  __shared__ float s_w[16 * 3];                       // attention weights
  __shared__ float s_last[16 * F_IN];                 // last cbc / running pred
  __shared__ float s_mu[16], s_rv[16];                // layernorm stats

  _Float16* s_y0  = (_Float16*)s_union;                  // [3][16][256], phase A only
  _Float16* s_din = (_Float16*)s_union;                  // [16][416], 13312 B
  float*    s_z   = (float*)(s_union + 13312);           // [16][128] f32, baseline phase
  _Float16* s_r   = (_Float16*)(s_union + 13312);        // [16][128] f16, decoder head

  const int tid  = threadIdx.x;
  const int lane = tid & 31;
  const int wave = tid >> 5;
  const int nb   = wave * 32;            // this wave's hidden-column base
  const int b0   = blockIdx.x * 16;      // batch-tile base

  float c_reg[2][8];
#pragma unroll
  for (int cc = 0; cc < 2; ++cc)
#pragma unroll
    for (int v = 0; v < 8; ++v) c_reg[cc][v] = 0.f;

  // ---- init: zero h, stage padded input tiles, seed last_cbc ----
  for (int i = tid; i < 16 * H; i += 256) s_h[i] = (_Float16)0.f;
  for (int i = tid; i < 3 * 16 * 32; i += 256) {
    int t = i >> 9, rem = i & 511, r = rem >> 5, k = rem & 31;
    s_x[i] = (k < F_IN)
                 ? (_Float16)x[(size_t)(b0 + r) * (T_IN * F_IN) + t * F_IN + k]
                 : (_Float16)0.f;
  }
  if (tid < 16 * F_IN) {
    int r = tid / F_IN, f = tid - r * F_IN;
    s_last[tid] = x[(size_t)(b0 + r) * (T_IN * F_IN) + 2 * F_IN + f];
  }
  __syncthreads();

  // ================= encoder layer 0 =================
  for (int t = 0; t < T_IN; ++t) {
    opaque_iter();                       // no cross-step weight caching -> no spills
    v8f acc[2][4] = {};
    {
      v16h a = load_a_lds(s_x + t * 512, 32, 0, lane);
#pragma unroll
      for (int cc = 0; cc < 2; ++cc)
#pragma unroll
        for (int g = 0; g < 4; ++g) {
          v16h b = load_b_glb(ws + OFF_IH0, G4H, 0, g * H + nb + cc * 16, lane);
          acc[cc][g] = __builtin_amdgcn_wmma_f32_16x16x32_f16(
              false, a, false, b, (short)0, acc[cc][g], false, false);
        }
    }
    gemm_gates(acc, s_h, H, ws + OFF_HH0, nb, lane);
    __syncthreads();
    lstm_elem(acc, c_reg, nb, lane, b_ih0, b_hh0, s_h, s_y0 + t * 16 * H);
    __syncthreads();
  }

  // ================= encoder layer 1 =================
  for (int i = tid; i < 16 * H; i += 256) s_h[i] = (_Float16)0.f;
#pragma unroll
  for (int cc = 0; cc < 2; ++cc)
#pragma unroll
    for (int v = 0; v < 8; ++v) c_reg[cc][v] = 0.f;
  __syncthreads();

  for (int t = 0; t < T_IN; ++t) {
    opaque_iter();
    v8f acc[2][4] = {};
    gemm_gates(acc, s_y0 + t * 16 * H, H, ws + OFF_IH1, nb, lane);
    gemm_gates(acc, s_h, H, ws + OFF_HH1, nb, lane);
    __syncthreads();
    lstm_elem(acc, c_reg, nb, lane, b_ih1, b_hh1, s_h, s_y1 + t * 16 * H);
    __syncthreads();
  }
  // c_reg now holds cT, s_h holds hT (decoder initial state)

  // ================= baseline branch (tiny, VALU) =================
  for (int i = tid; i < 16 * BD; i += 256) {
    int r = i >> 7, c = i & 127;
    float a = bb[c];
#pragma unroll
    for (int k = 0; k < NBASE; ++k)
      a += baseline[(size_t)(b0 + r) * NBASE + k] * Wb[c * NBASE + k];
    s_z[i] = fmaxf(a, 0.f);
  }
  __syncthreads();
  if (tid < 16) {
    float mu = 0.f;
    for (int k = 0; k < BD; ++k) mu += s_z[tid * BD + k];
    mu *= (1.f / BD);
    float var = 0.f;
    for (int k = 0; k < BD; ++k) { float d = s_z[tid * BD + k] - mu; var += d * d; }
    var *= (1.f / BD);
    s_mu[tid] = mu;
    s_rv[tid] = rsqrtf(var + 1e-5f);
  }
  __syncthreads();
  // baseline_repr directly into dec_in columns [262,390)
  for (int i = tid; i < 16 * BD; i += 256) {
    int r = i >> 7, c = i & 127;
    float v = (s_z[i] - s_mu[r]) * s_rv[r] * ln_g[c] + ln_b[c];
    s_din[r * DINP + 262 + c] = (_Float16)v;
  }
  // zero-pad dec_in columns [390,416) and seed columns [0,6)
  for (int i = tid; i < 16 * (DINP - DIN); i += 256) {
    int r = i / (DINP - DIN), c = i - r * (DINP - DIN);
    s_din[r * DINP + DIN + c] = (_Float16)0.f;
  }
  if (tid < 16 * F_IN) {
    int r = tid / F_IN, f = tid - r * F_IN;
    s_din[r * DINP + f] = (_Float16)s_last[tid];
  }
  __syncthreads();

  // ================= decoder: 5 autoregressive steps =================
  for (int step = 0; step < T_OUT; ++step) {
    opaque_iter();                       // reload weights from L2 each step
    // ---- attention: fused q-proj + enc-proj + tanh dot-with-va ----
    if (tid < 48) s_sc[tid] = 0.f;
    __syncthreads();
    {
      // every fragment loaded exactly once: accumulate q and enc-proj together
      v8f qacc[2] = {};
      v8f eacc[2][3] = {};
      for (int kc = 0; kc < H / 32; ++kc) {
        v16h ah = load_a_lds(s_h, H, kc, lane);
        v16h a0 = load_a_lds(s_y1 + 0 * 16 * H, H, kc, lane);
        v16h a1 = load_a_lds(s_y1 + 1 * 16 * H, H, kc, lane);
        v16h a2 = load_a_lds(s_y1 + 2 * 16 * H, H, kc, lane);
#pragma unroll
        for (int cc = 0; cc < 2; ++cc) {
          v16h bq = load_b_glb(ws + OFF_1A, H, kc, nb + cc * 16, lane);
          qacc[cc] = __builtin_amdgcn_wmma_f32_16x16x32_f16(
              false, ah, false, bq, (short)0, qacc[cc], false, false);
          v16h be = load_b_glb(ws + OFF_2A, H, kc, nb + cc * 16, lane);
          eacc[cc][0] = __builtin_amdgcn_wmma_f32_16x16x32_f16(
              false, a0, false, be, (short)0, eacc[cc][0], false, false);
          eacc[cc][1] = __builtin_amdgcn_wmma_f32_16x16x32_f16(
              false, a1, false, be, (short)0, eacc[cc][1], false, false);
          eacc[cc][2] = __builtin_amdgcn_wmma_f32_16x16x32_f16(
              false, a2, false, be, (short)0, eacc[cc][2], false, false);
        }
      }
      // epilogue: scores += tanh(q + ep + (b1a+b2a)) * va, reduced via LDS atomics
      float sacc[3][8];
#pragma unroll
      for (int t = 0; t < 3; ++t)
#pragma unroll
        for (int v = 0; v < 8; ++v) sacc[t][v] = 0.f;
#pragma unroll
      for (int cc = 0; cc < 2; ++cc) {
        const int col = nb + cc * 16 + (lane & 15);
        const float qb  = b1a[col] + b2a[col];
        const float vav = va[col];
#pragma unroll
        for (int t = 0; t < 3; ++t)
#pragma unroll
          for (int v = 0; v < 8; ++v)
            sacc[t][v] += tanhf(qacc[cc][v] + eacc[cc][t][v] + qb) * vav;
      }
      const int hi = (lane >> 4) << 3;
#pragma unroll
      for (int t = 0; t < 3; ++t)
#pragma unroll
        for (int v = 0; v < 8; ++v)
          atomicAdd(&s_sc[(hi + v) * 3 + t], sacc[t][v]);
    }
    __syncthreads();

    // ---- softmax over 3 timesteps (ba cancels in softmax) ----
    if (tid < 16) {
      float s0 = s_sc[tid * 3], s1 = s_sc[tid * 3 + 1], s2 = s_sc[tid * 3 + 2];
      float m = fmaxf(s0, fmaxf(s1, s2));
      float e0 = __expf(s0 - m), e1 = __expf(s1 - m), e2 = __expf(s2 - m);
      float inv = 1.f / (e0 + e1 + e2);
      s_w[tid * 3] = e0 * inv; s_w[tid * 3 + 1] = e1 * inv; s_w[tid * 3 + 2] = e2 * inv;
    }
    __syncthreads();

    // ---- context -> dec_in columns [6,262) ----
    for (int i = tid; i < 16 * H; i += 256) {
      int r = i >> 8, c = i & 255;
      float ctx = s_w[r * 3 + 0] * (float)s_y1[0 * 16 * H + r * H + c]
                + s_w[r * 3 + 1] * (float)s_y1[1 * 16 * H + r * H + c]
                + s_w[r * 3 + 2] * (float)s_y1[2 * 16 * H + r * H + c];
      s_din[r * DINP + F_IN + c] = (_Float16)ctx;
    }
    __syncthreads();

    // ---- decoder LSTM cell ----
    {
      v8f acc[2][4] = {};
      gemm_gates(acc, s_din, DINP, ws + OFF_DIH, nb, lane);  // 13 k-chunks
      gemm_gates(acc, s_h, H, ws + OFF_DHH, nb, lane);       //  8 k-chunks
      __syncthreads();
      lstm_elem(acc, c_reg, nb, lane, bd_ih, bd_hh, s_h, nullptr);
    }
    __syncthreads();

    // ---- head: relu(h @ Wo1^T + bo1) via WMMA (128 cols / 8 waves) ----
    {
      v8f oa = {};
      const int n0 = wave * 16;
      for (int kc = 0; kc < H / 32; ++kc) {
        v16h a = load_a_lds(s_h, H, kc, lane);
        v16h b = load_b_glb(ws + OFF_O1, BD, kc, n0, lane);
        oa = __builtin_amdgcn_wmma_f32_16x16x32_f16(
            false, a, false, b, (short)0, oa, false, false);
      }
      const int col = n0 + (lane & 15);
      const int hi = (lane >> 4) << 3;
      const float bv = bo1[col];
#pragma unroll
      for (int v = 0; v < 8; ++v)
        s_r[(hi + v) * BD + col] = (_Float16)fmaxf(oa[v] + bv, 0.f);
    }
    __syncthreads();

    // ---- delta = relu_buf @ Wo2^T + bo2 ; pred = last + delta ----
    if (tid < 16 * F_IN) {
      int r = tid / F_IN, f = tid - r * F_IN;
      float d = bo2[f];
      for (int k = 0; k < BD; ++k) d += (float)s_r[r * BD + k] * Wo2[f * BD + k];
      float pred = s_last[tid] + d;
      out[(size_t)(b0 + r) * (T_OUT * F_IN) + step * F_IN + f] = pred;
      s_last[tid] = pred;
      s_din[r * DINP + f] = (_Float16)pred;
    }
    __syncthreads();
  }
}

// ---------------- host-side launch ----------------
extern "C" void kernel_launch(void* const* d_in, const int* in_sizes, int n_in,
                              void* d_out, int out_size, void* d_ws, size_t ws_size,
                              hipStream_t stream) {
  (void)in_sizes; (void)n_in; (void)out_size; (void)ws_size;
  _Float16* ws = (_Float16*)d_ws;

  struct PrepJob { int in_idx; size_t off; int N, K, Kpad; };
  const PrepJob jobs[9] = {
      { 2, OFF_IH0, 1024,   6,  32},   // W_ih0
      { 3, OFF_HH0, 1024, 256, 256},   // W_hh0
      { 6, OFF_IH1, 1024, 256, 256},   // W_ih1
      { 7, OFF_HH1, 1024, 256, 256},   // W_hh1
      {16, OFF_2A,   256, 256, 256},   // W2a
      {14, OFF_1A,   256, 256, 256},   // W1a
      {20, OFF_DIH, 1024, 390, 416},   // Wd_ih
      {21, OFF_DHH, 1024, 256, 256},   // Wd_hh
      {24, OFF_O1,   128, 256, 256},   // Wo1
  };
  for (int j = 0; j < 9; ++j) {
    int total = jobs[j].Kpad * jobs[j].N;
    prep_wt_kernel<<<(total + 255) / 256, 256, 0, stream>>>(
        (const float*)d_in[jobs[j].in_idx], ws + jobs[j].off,
        jobs[j].N, jobs[j].K, jobs[j].Kpad);
  }

  cbc_forecast_kernel<<<BTOT / 16, 256, 0, stream>>>(
      (const float*)d_in[0],  (const float*)d_in[1],
      (const float*)d_in[4],  (const float*)d_in[5],
      (const float*)d_in[8],  (const float*)d_in[9],
      (const float*)d_in[10], (const float*)d_in[11],
      (const float*)d_in[12], (const float*)d_in[13],
      (const float*)d_in[15], (const float*)d_in[17],
      (const float*)d_in[18],
      (const float*)d_in[22], (const float*)d_in[23],
      (const float*)d_in[25], (const float*)d_in[26],
      (const float*)d_in[27],
      (const _Float16*)ws, (float*)d_out);
}
